// SetCoverHolo_46806553592240
// MI455X (gfx1250) — matrix operand order repaired and checked by reading, not compiled, not run on hardware
//
#include <hip/hip_runtime.h>
#include <hip/hip_bf16.h>

#define DD 64
#define WPB 8            // waves per block in the edge kernel (256 threads)
#define EDGE_BLOCKS 4096
#define PADW 68          // LDS msg row stride (floats): 16B-aligned, bank-staggered
#define WTP 72           // LDS W_fin^T row stride (bf16): 16B-aligned, bank-staggered

typedef __attribute__((ext_vector_type(16))) __bf16 v16bf;
typedef __attribute__((ext_vector_type(8)))  __bf16 v8bf;
typedef __attribute__((ext_vector_type(8)))  float  v8f;

static __device__ __forceinline__ unsigned pack2bf(float lo, float hi)
{
    const unsigned short a = __builtin_bit_cast(unsigned short, (__bf16)lo);
    const unsigned short b = __builtin_bit_cast(unsigned short, (__bf16)hi);
    return (unsigned)a | ((unsigned)b << 16);
}

// ---------------------------------------------------------------------------
// XL = X @ W (+ b)   for [n,64] x [64,64] node-side linear layers
// ---------------------------------------------------------------------------
__global__ __launch_bounds__(256) void node_linear_kern(
    const float* __restrict__ X, const float* __restrict__ W,
    const float* __restrict__ b, float* __restrict__ Y, int n)
{
    __shared__ float sx[4][DD];
    const int sub = threadIdx.x >> 6;
    const int col = threadIdx.x & 63;
    const int row = blockIdx.x * 4 + sub;
    if (row < n) sx[sub][col] = X[(size_t)row * DD + col];
    __syncthreads();
    if (row < n) {
        float acc = b ? b[col] : 0.f;
        #pragma unroll
        for (int k = 0; k < DD; ++k) acc += sx[sub][k] * W[k * DD + col];
        Y[(size_t)row * DD + col] = acc;
    }
}

// ---------------------------------------------------------------------------
// Fused edge kernel: msg = XL[dst] + XR[src] + ef*W_edge ; LN ; ReLU ;
//                    (msg @ W_fin + bb_fin) scatter-added into agg[dst]
// One wave handles 16 edges/tile; matmul = 8x v_wmma_f32_16x16x32_bf16.
// Invalid edges are routed to dummy row n_right (no divergent atomics).
// ---------------------------------------------------------------------------
__global__ __launch_bounds__(256) void edge_wmma_kern(
    const int* __restrict__ src_idx, const int* __restrict__ dst_idx,
    const float* __restrict__ ef,
    const float* __restrict__ XL, const float* __restrict__ XR,
    const float* __restrict__ W_edge,
    const float* __restrict__ g_fin, const float* __restrict__ b_fin,
    const float* __restrict__ W_fin, const float* __restrict__ bb_fin,
    float* __restrict__ agg, int n_edges, int dummy_row)
{
    // LDS pool: first used to stage W_fin^T (bf16), then reused as msg tiles.
    __shared__ __align__(16) float s_pool[WPB * 16 * PADW];   // 34816 B
    __shared__ int   s_dst[WPB][16];
    __shared__ int   s_src[WPB][16];
    __shared__ float s_ef [WPB][16];

    float          (*s_msg)[16][PADW] = (float (*)[16][PADW])s_pool;
    unsigned short (*s_wt)[WTP]       = (unsigned short (*)[WTP])s_pool; // [64][72]

    const int  wave = threadIdx.x >> 5;
    const int  lane = threadIdx.x & 31;
    const int  Mrow = lane & 15;
    const bool hi   = lane >= 16;
    const int  kb   = hi ? 8 : 0;      // this lane's A-fragment K base
    const int  q4   = Mrow * 4;        // this lane's fixed float4 column group

    // ---- cooperative staging: W_fin^T (n-major, k-contiguous) as bf16 ----
    {
        const int t  = threadIdx.x;          // 0..255
        const int k0 = (t >> 3) * 2;         // even k: 0..62
        const int n0 = (t & 7) * 8;          // 0..56
        const float4 a0 = *(const float4*)(W_fin + k0 * DD + n0);
        const float4 a1 = *(const float4*)(W_fin + k0 * DD + n0 + 4);
        const float4 c0 = *(const float4*)(W_fin + (k0 + 1) * DD + n0);
        const float4 c1 = *(const float4*)(W_fin + (k0 + 1) * DD + n0 + 4);
        const float va[8] = {a0.x, a0.y, a0.z, a0.w, a1.x, a1.y, a1.z, a1.w};
        const float vb[8] = {c0.x, c0.y, c0.z, c0.w, c1.x, c1.y, c1.z, c1.w};
        #pragma unroll
        for (int i = 0; i < 8; ++i)
            *(unsigned*)&s_wt[n0 + i][k0] = pack2bf(va[i], vb[i]);
    }
    __syncthreads();

    // ---- per-wave invariants in registers -------------------------------
    // B fragments of W_fin (K=64 x N=64), bf16; 2 K-steps x 4 N-tiles.
    v16bf B[2][4];
    #pragma unroll
    for (int kt = 0; kt < 2; ++kt) {
        #pragma unroll
        for (int nt = 0; nt < 4; ++nt) {
            const int kk = kt * 32 + (hi ? 8 : 0);
            const int n  = nt * 16 + Mrow;
            const v8bf blo = *(const v8bf*)(&s_wt[n][kk]);
            const v8bf bhi = *(const v8bf*)(&s_wt[n][kk + 16]);
            #pragma unroll
            for (int i = 0; i < 8; ++i) {
                B[kt][nt][i]     = blo[i];
                B[kt][nt][8 + i] = bhi[i];
            }
        }
    }
    __syncthreads();   // staging lifetime ends; pool becomes s_msg

    const float4 we4 = *(const float4*)(W_edge + q4);
    float4 ga[4], gb[4], ba[4], bb4[4];
    #pragma unroll
    for (int g = 0; g < 4; ++g) {
        const int k0 = kb + g * 16;
        ga[g]  = *(const float4*)(g_fin + k0);
        gb[g]  = *(const float4*)(g_fin + k0 + 4);
        ba[g]  = *(const float4*)(b_fin + k0);
        bb4[g] = *(const float4*)(b_fin + k0 + 4);
    }
    float bbv[4];
    #pragma unroll
    for (int nt = 0; nt < 4; ++nt) bbv[nt] = bb_fin[nt * 16 + Mrow];

    const int tiles        = (n_edges + 15) >> 4;
    const int tilesPerIter = gridDim.x * WPB;
    const int nIter        = (tiles + tilesPerIter - 1) / tilesPerIter;

    for (int it = 0; it < nIter; ++it) {
        const int  tile   = it * tilesPerIter + blockIdx.x * WPB + wave;
        const bool active = tile < tiles;            // wave-uniform
        const int  e0     = tile << 4;

        if (lane < 16) {
            int dd = dummy_row, ss = dummy_row; float fv = 0.f;
            const int e = e0 + lane;
            if (active && e < n_edges) { dd = dst_idx[e]; ss = src_idx[e]; fv = ef[e]; }
            s_dst[wave][lane] = dd; s_src[wave][lane] = ss; s_ef[wave][lane] = fv;
        }
        __syncthreads();

        // ---- build 16x64 message tile: float4 gathers, b128 LDS stores ---
        #pragma unroll
        for (int j = 0; j < 8; ++j) {
            const int er = 2 * j + (hi ? 1 : 0);     // half-wave per row
            const int dd = s_dst[wave][er];
            const int ss = s_src[wave][er];
            const float fe = s_ef[wave][er];
            const float4 xl = *(const float4*)(XL + (unsigned)(dd * DD) + q4);
            const float4 xr = *(const float4*)(XR + (unsigned)(ss * DD) + q4);
            float4 v;
            v.x = xl.x + xr.x + fe * we4.x;
            v.y = xl.y + xr.y + fe * we4.y;
            v.z = xl.z + xr.z + fe * we4.z;
            v.w = xl.w + xr.w + fe * we4.w;
            *(float4*)(&s_msg[wave][er][q4]) = v;
        }
        __syncthreads();

        // ---- LayerNorm stats: half-row per lane + one cross-lane combine -
        float s = 0.f, s2 = 0.f;
        #pragma unroll
        for (int t = 0; t < 8; ++t) {
            const float4 x = *(const float4*)(&s_msg[wave][Mrow][(hi ? 32 : 0) + t * 4]);
            s  += x.x + x.y + x.z + x.w;
            s2 += x.x * x.x + x.y * x.y + x.z * x.z + x.w * x.w;
        }
        s  += __shfl_xor(s, 16);
        s2 += __shfl_xor(s2, 16);
        const float mu  = s * (1.f / DD);
        const float var = s2 * (1.f / DD) - mu * mu;
        const float rs  = rsqrtf(var + 1e-5f);

        // ---- pack A fragments (LN + ReLU + bf16), b128 LDS reads ---------
        v16bf A0, A1;
        #pragma unroll
        for (int g = 0; g < 4; ++g) {
            const int k0 = kb + g * 16;
            const float4 lo = *(const float4*)(&s_msg[wave][Mrow][k0]);
            const float4 hp = *(const float4*)(&s_msg[wave][Mrow][k0 + 4]);
            const float e0v = fmaxf((lo.x - mu) * rs * ga[g].x + ba[g].x, 0.f);
            const float e1v = fmaxf((lo.y - mu) * rs * ga[g].y + ba[g].y, 0.f);
            const float e2v = fmaxf((lo.z - mu) * rs * ga[g].z + ba[g].z, 0.f);
            const float e3v = fmaxf((lo.w - mu) * rs * ga[g].w + ba[g].w, 0.f);
            const float e4v = fmaxf((hp.x - mu) * rs * gb[g].x + bb4[g].x, 0.f);
            const float e5v = fmaxf((hp.y - mu) * rs * gb[g].y + bb4[g].y, 0.f);
            const float e6v = fmaxf((hp.z - mu) * rs * gb[g].z + bb4[g].z, 0.f);
            const float e7v = fmaxf((hp.w - mu) * rs * gb[g].w + bb4[g].w, 0.f);
            const int bidx = (g & 1) * 8;
            if (g < 2) {
                A0[bidx + 0] = (__bf16)e0v; A0[bidx + 1] = (__bf16)e1v;
                A0[bidx + 2] = (__bf16)e2v; A0[bidx + 3] = (__bf16)e3v;
                A0[bidx + 4] = (__bf16)e4v; A0[bidx + 5] = (__bf16)e5v;
                A0[bidx + 6] = (__bf16)e6v; A0[bidx + 7] = (__bf16)e7v;
            } else {
                A1[bidx + 0] = (__bf16)e0v; A1[bidx + 1] = (__bf16)e1v;
                A1[bidx + 2] = (__bf16)e2v; A1[bidx + 3] = (__bf16)e3v;
                A1[bidx + 4] = (__bf16)e4v; A1[bidx + 5] = (__bf16)e5v;
                A1[bidx + 6] = (__bf16)e6v; A1[bidx + 7] = (__bf16)e7v;
            }
        }

        // ---- [16,64] x [64,64]: 2 K-steps x 4 N-tiles --------------------
        v8f C0 = {}, C1 = {}, C2 = {}, C3 = {};
        C0 = __builtin_amdgcn_wmma_f32_16x16x32_bf16(false, A0, false, B[0][0], (short)0, C0, false, false);
        C1 = __builtin_amdgcn_wmma_f32_16x16x32_bf16(false, A0, false, B[0][1], (short)0, C1, false, false);
        C2 = __builtin_amdgcn_wmma_f32_16x16x32_bf16(false, A0, false, B[0][2], (short)0, C2, false, false);
        C3 = __builtin_amdgcn_wmma_f32_16x16x32_bf16(false, A0, false, B[0][3], (short)0, C3, false, false);
        C0 = __builtin_amdgcn_wmma_f32_16x16x32_bf16(false, A1, false, B[1][0], (short)0, C0, false, false);
        C1 = __builtin_amdgcn_wmma_f32_16x16x32_bf16(false, A1, false, B[1][1], (short)0, C1, false, false);
        C2 = __builtin_amdgcn_wmma_f32_16x16x32_bf16(false, A1, false, B[1][2], (short)0, C2, false, false);
        C3 = __builtin_amdgcn_wmma_f32_16x16x32_bf16(false, A1, false, B[1][3], (short)0, C3, false, false);

        // ---- scatter: 8 offsets once, +nt*16 folds into inst offset ------
        if (active) {                                 // wave-uniform branch
            unsigned voff[8];
            #pragma unroll
            for (int j = 0; j < 8; ++j)
                voff[j] = (unsigned)(s_dst[wave][hi ? j + 8 : j] * DD + Mrow);
            #pragma unroll
            for (int nt = 0; nt < 4; ++nt) {
                const float bb = bbv[nt];
                const v8f   cc = (nt == 0) ? C0 : (nt == 1) ? C1 : (nt == 2) ? C2 : C3;
                #pragma unroll
                for (int j = 0; j < 8; ++j)
                    unsafeAtomicAdd(agg + voff[j] + nt * 16, cc[j] + bb);
            }
        }
    }
}

// ---------------------------------------------------------------------------
// Per-node output: h = [LN(agg); R]; R += relu(h@W_o1+b_o1)@W_o2 + b_o2
// ---------------------------------------------------------------------------
__global__ __launch_bounds__(256) void node_out_kern(
    const float* __restrict__ agg, float* __restrict__ R,
    const float* __restrict__ g_post, const float* __restrict__ b_post,
    const float* __restrict__ W_o1, const float* __restrict__ b_o1,
    const float* __restrict__ W_o2, const float* __restrict__ b_o2, int n)
{
    __shared__ float sh[4][2 * DD];
    __shared__ float st[4][DD];
    __shared__ float sstat[4][2];
    const int sub = threadIdx.x >> 6;
    const int col = threadIdx.x & 63;
    const int row = blockIdx.x * 4 + sub;
    const bool ok = row < n;

    const float a  = ok ? agg[(size_t)row * DD + col] : 0.f;
    const float rv = ok ? R[(size_t)row * DD + col] : 0.f;
    sh[sub][col] = a;
    __syncthreads();
    if (col == 0) {
        float s = 0.f, s2 = 0.f;
        #pragma unroll
        for (int k = 0; k < DD; ++k) { const float x = sh[sub][k]; s += x; s2 += x * x; }
        const float mu  = s * (1.f / DD);
        const float var = s2 * (1.f / DD) - mu * mu;
        sstat[sub][0] = mu;
        sstat[sub][1] = rsqrtf(var + 1e-5f);
    }
    __syncthreads();
    const float mu = sstat[sub][0], rs = sstat[sub][1];
    sh[sub][col]      = (a - mu) * rs * g_post[col] + b_post[col];
    sh[sub][DD + col] = rv;
    __syncthreads();
    float t = b_o1[col];
    #pragma unroll
    for (int k = 0; k < 2 * DD; ++k) t += sh[sub][k] * W_o1[k * DD + col];
    st[sub][col] = fmaxf(t, 0.f);
    __syncthreads();
    float o = b_o2[col];
    #pragma unroll
    for (int k = 0; k < DD; ++k) o += st[sub][k] * W_o2[k * DD + col];
    if (ok) R[(size_t)row * DD + col] = rv + o;
}

// ---------------------------------------------------------------------------
// Host side
// ---------------------------------------------------------------------------
struct ConvP {
    const float *W_left, *b_left, *W_edge, *W_right, *g_fin, *b_fin, *W_fin,
                *bb_fin, *g_post, *b_post, *W_o1, *b_o1, *W_o2, *b_o2;
};

static ConvP get_conv(void* const* d_in, int conv)
{
    // params flattened in dict insertion order:
    //  conv_0_v_to_c, conv_0_c_to_v, conv_1_v_to_c, conv_1_c_to_v,
    //  each with 14 tensors in _conv_params() insertion order.
    ConvP p;
    const float** f = (const float**)&p;
    const int base = 4 + conv * 14;
    for (int i = 0; i < 14; ++i) f[i] = (const float*)d_in[base + i];
    return p;
}

static void run_conv(const float* left, int n_left,
                     const int* src, const int* dst,
                     float* right, int n_right,
                     const float* ef, int n_edges, const ConvP& p,
                     float* XL, float* XR, float* agg, hipStream_t stream)
{
    node_linear_kern<<<(n_right + 3) / 4, 256, 0, stream>>>(right, p.W_left, p.b_left, XL, n_right);
    node_linear_kern<<<(n_left  + 3) / 4, 256, 0, stream>>>(left,  p.W_right, nullptr, XR, n_left);
    hipMemsetAsync(agg, 0, (size_t)n_right * DD * sizeof(float), stream);
    edge_wmma_kern<<<EDGE_BLOCKS, 32 * WPB, 0, stream>>>(
        src, dst, ef, XL, XR, p.W_edge, p.g_fin, p.b_fin, p.W_fin, p.bb_fin,
        agg, n_edges, /*dummy_row=*/n_right);
    node_out_kern<<<(n_right + 3) / 4, 256, 0, stream>>>(
        agg, right, p.g_post, p.b_post, p.W_o1, p.b_o1, p.W_o2, p.b_o2, n_right);
}

extern "C" void kernel_launch(void* const* d_in, const int* in_sizes, int n_in,
                              void* d_out, int out_size, void* d_ws, size_t ws_size,
                              hipStream_t stream)
{
    const float* consF = (const float*)d_in[0];
    const int*   eidx  = (const int*)d_in[1];
    const float* ef    = (const float*)d_in[2];
    const float* varF  = (const float*)d_in[3];

    const int n_cons  = in_sizes[0] / DD;
    const int n_edges = in_sizes[2];          // EDGE_NF == 1
    const int n_vars  = in_sizes[3] / DD;

    const int* cons_idx = eidx;
    const int* var_idx  = eidx + n_edges;

    // Live node states live directly in d_out: [C (n_cons x 64) ; V (n_vars x 64)]
    float* C = (float*)d_out;
    float* V = C + (size_t)n_cons * DD;
    hipMemcpyAsync(C, consF, (size_t)n_cons * DD * sizeof(float), hipMemcpyDeviceToDevice, stream);
    hipMemcpyAsync(V, varF,  (size_t)n_vars * DD * sizeof(float), hipMemcpyDeviceToDevice, stream);

    // Workspace: XL, XR, agg — each with 16 rows of slack for the dummy row
    const size_t stride = (size_t)(n_vars + 16) * DD;
    float* XL  = (float*)d_ws;
    float* XR  = XL + stride;
    float* agg = XR + stride;

    for (int layer = 0; layer < 2; ++layer) {
        const ConvP pvc = get_conv(d_in, layer * 2 + 0);   // conv_{l}_v_to_c
        const ConvP pcv = get_conv(d_in, layer * 2 + 1);   // conv_{l}_c_to_v
        // v_to_c: left=V (src=var), right=C (dst=cons)
        run_conv(V, n_vars, var_idx, cons_idx, C, n_cons, ef, n_edges, pvc, XL, XR, agg, stream);
        // c_to_v: left=C (src=cons), right=V (dst=var)
        run_conv(C, n_cons, cons_idx, var_idx, V, n_vars, ef, n_edges, pcv, XL, XR, agg, stream);
    }
}
